// When2commFusion_13932873908853
// MI455X (gfx1250) — matrix operand order repaired and checked by reading, not compiled, not run on hardware
//
#include <hip/hip_runtime.h>
#include <hip/hip_bf16.h>

typedef __attribute__((ext_vector_type(16))) _Float16 v16h;
typedef __attribute__((ext_vector_type(8)))  _Float16 v8h;
typedef __attribute__((ext_vector_type(4)))  _Float16 v4h;
typedef __attribute__((ext_vector_type(8)))  float    v8f;

constexpr int B_ = 4, L_ = 5, C_ = 64, H_ = 256, W_ = 128;
constexpr int NIMG = B_ * L_;   // 20

// ---------------------------------------------------------------------------
// Kernel 1: affine grid + bilinear sample; write warped maps as f16.
// One thread per (b, n, h, w); loops channels (64).
// ---------------------------------------------------------------------------
__global__ void __launch_bounds__(256)
grid_sample_f16(const float* __restrict__ x, const float* __restrict__ nam,
                _Float16* __restrict__ neigh) {
    int tid = blockIdx.x * 256 + threadIdx.x;          // B*L*H*W = 655360
    int w = tid & (W_ - 1);
    int t = tid >> 7;
    int h = t & (H_ - 1);
    t >>= 8;
    int n = t % L_;
    int b = t / L_;
    if (b >= B_) return;

    const float* th = nam + ((size_t)(b * L_ * L_) + n) * 6;  // nam[b,0,n]
    float gx = (w + 0.5f) * (2.0f / W_) - 1.0f;
    float gy = (h + 0.5f) * (2.0f / H_) - 1.0f;
    float sx = th[0] * gx + th[1] * gy + th[2];
    float sy = th[3] * gx + th[4] * gy + th[5];
    float fx = ((sx + 1.0f) * W_ - 1.0f) * 0.5f;
    float fy = ((sy + 1.0f) * H_ - 1.0f) * 0.5f;
    float x0f = floorf(fx), y0f = floorf(fy);
    int x0 = (int)x0f, y0 = (int)y0f;
    float wx1 = fx - x0f, wx0 = 1.0f - wx1;
    float wy1 = fy - y0f, wy0 = 1.0f - wy1;

    int x1 = x0 + 1, y1 = y0 + 1;
    float m00 = (x0 >= 0 && x0 < W_ && y0 >= 0 && y0 < H_) ? wx0 * wy0 : 0.0f;
    float m10 = (x1 >= 0 && x1 < W_ && y0 >= 0 && y0 < H_) ? wx1 * wy0 : 0.0f;
    float m01 = (x0 >= 0 && x0 < W_ && y1 >= 0 && y1 < H_) ? wx0 * wy1 : 0.0f;
    float m11 = (x1 >= 0 && x1 < W_ && y1 >= 0 && y1 < H_) ? wx1 * wy1 : 0.0f;
    int xc0 = x0 < 0 ? 0 : (x0 > W_ - 1 ? W_ - 1 : x0);
    int xc1 = x1 < 0 ? 0 : (x1 > W_ - 1 ? W_ - 1 : x1);
    int yc0 = y0 < 0 ? 0 : (y0 > H_ - 1 ? H_ - 1 : y0);
    int yc1 = y1 < 0 ? 0 : (y1 > H_ - 1 ? H_ - 1 : y1);

    const float* img = x + (size_t)(b * L_ + n) * C_ * H_ * W_;
    _Float16* dst = neigh + (size_t)(b * L_ + n) * C_ * H_ * W_ + (size_t)h * W_ + w;
    int i00 = yc0 * W_ + xc0, i10 = yc0 * W_ + xc1;
    int i01 = yc1 * W_ + xc0, i11 = yc1 * W_ + xc1;
    for (int c = 0; c < C_; ++c) {
        const float* p = img + (size_t)c * H_ * W_;
        float v = m00 * p[i00] + m10 * p[i10] + m01 * p[i01] + m11 * p[i11];
        dst[(size_t)c * H_ * W_] = (_Float16)v;
    }
}

// ---------------------------------------------------------------------------
// Kernel 2: 4x4 stride-2 pad-1 conv + bias + ReLU via implicit-GEMM WMMA.
// 8 waves (256 thr) per block; block tile = 32(Cout) x 64(spatial); wave wv
// owns sub-tile (mt = wv&1, nt = wv>>1). K = CIN*16 staged in LDS f16 in
// 256-chunks, x4-vectorized (ds_store_b64), consumed by
// v_wmma_f32_16x16x32_f16 with the ISA fragment layouts.
// ---------------------------------------------------------------------------
template <int CIN, int COUT, int HIN, int WIN>
__global__ void __launch_bounds__(256)
conv4x4s2_wmma(const _Float16* __restrict__ in, const float* __restrict__ wgt,
               const float* __restrict__ bias, _Float16* __restrict__ outp) {
    constexpr int HOUT = HIN / 2, WOUT = WIN / 2;
    constexpr int K = CIN * 16;
    constexpr int KC = 256;                 // K-chunk
    constexpr int KG = KC / 4;              // 64 quad-groups per row

    __shared__ __align__(16) _Float16 Alds[32 * KC];   // 16 KB
    __shared__ __align__(16) _Float16 Blds[64 * KC];   // 32 KB

    const int t    = threadIdx.x;
    const int img  = blockIdx.z;
    const int cm0  = blockIdx.y * 32;
    const int s0   = blockIdx.x * 64;
    const _Float16* inImg = in + (size_t)img * CIN * HIN * WIN;

    const int lane = t & 31;
    const int lm = lane & 15;
    const int lh = lane >> 4;
    const int wv = t >> 5;
    const int mt = wv & 1;                  // cout sub-tile
    const int nt = wv >> 1;                 // spatial sub-tile
    v8f acc = {};

    for (int kc0 = 0; kc0 < K; kc0 += KC) {
        // ---- Stage A: 32 x KC weight slice, fp32 -> f16, x4 vectorized ----
        for (int g = t; g < 32 * KG; g += 256) {
            int m = g >> 6;                 // g / KG
            int k4 = (g & (KG - 1)) * 4;
            const float* wp = wgt + (size_t)(cm0 + m) * K + kc0 + k4;
            float4 wv4 = *(const float4*)wp;
            v4h hq = { (_Float16)wv4.x, (_Float16)wv4.y,
                       (_Float16)wv4.z, (_Float16)wv4.w };
            *(v4h*)(Alds + m * KC + k4) = hq;
            if (kc0 + KC < K)               // prefetch next K-chunk of weights
                __builtin_prefetch(wp + KC, 0, 3);
        }
        // ---- Stage B: im2col 64 x KC, grouped by the 4 contiguous kw taps --
        for (int g = t; g < 64 * KG; g += 256) {
            int j  = g >> 6;                // spatial position in tile
            int k  = kc0 + (g & (KG - 1)) * 4;   // k multiple of 4 -> kw = 0..3
            int ci = k >> 4, kh = (k >> 2) & 3;
            int s  = s0 + j;
            int oh = s / WOUT, ow = s % WOUT;
            int ih = 2 * oh - 1 + kh;
            int iwb = 2 * ow - 1;
            bool hok = (ih >= 0) && (ih < HIN);
            int ihc = ih < 0 ? 0 : (ih > HIN - 1 ? HIN - 1 : ih);
            const _Float16* row = inImg + (size_t)ci * HIN * WIN + (size_t)ihc * WIN;
            v4h hq;
#pragma unroll
            for (int u = 0; u < 4; ++u) {
                int iw = iwb + u;
                bool ok = hok && (iw >= 0) && (iw < WIN);
                int iwc = iw < 0 ? 0 : (iw > WIN - 1 ? WIN - 1 : iw);
                _Float16 v = row[iwc];
                hq[u] = ok ? v : (_Float16)0.0f;
            }
            *(v4h*)(Blds + j * KC + (k - kc0)) = hq;
        }
        __syncthreads();

        // ---- WMMA: A 16x32 f16, B 32x16 f16, C f32 16x16 ----
        const _Float16* Ap = Alds + (mt * 16 + lm) * KC + lh * 8;
        const _Float16* Bp = Blds + (nt * 16 + lm) * KC + lh * 8;
#pragma unroll
        for (int kc = 0; kc < KC; kc += 32) {
            v8h a0 = *(const v8h*)(Ap + kc);
            v8h a1 = *(const v8h*)(Ap + kc + 16);
            v8h b0 = *(const v8h*)(Bp + kc);
            v8h b1 = *(const v8h*)(Bp + kc + 16);
            v16h a = __builtin_shufflevector(a0, a1, 0, 1, 2, 3, 4, 5, 6, 7,
                                             8, 9, 10, 11, 12, 13, 14, 15);
            v16h b = __builtin_shufflevector(b0, b1, 0, 1, 2, 3, 4, 5, 6, 7,
                                             8, 9, 10, 11, 12, 13, 14, 15);
            acc = __builtin_amdgcn_wmma_f32_16x16x32_f16(
                false, a, false, b, (short)0, acc, false, false);
        }
        __syncthreads();
    }

    // D layout: VGPR r -> M = r + 8*(lane>>4), N = lane&15.
    int sPos = s0 + nt * 16 + lm;
#pragma unroll
    for (int r = 0; r < 8; ++r) {
        int co = cm0 + mt * 16 + r + lh * 8;
        float v = acc[r] + bias[co];
        v = fmaxf(v, 0.0f);
        outp[(size_t)img * COUT * HOUT * WOUT + (size_t)co * HOUT * WOUT + sPos] =
            (_Float16)v;
    }
}

// ---------------------------------------------------------------------------
// Kernel 3: global mean pool of conv3 output (20 imgs x 64 ch over 32x16).
// ---------------------------------------------------------------------------
__global__ void __launch_bounds__(256)
pool_mean(const _Float16* __restrict__ o3, float* __restrict__ pooled) {
    int tid = blockIdx.x * 256 + threadIdx.x;   // 20*64 = 1280
    if (tid >= NIMG * 64) return;
    const _Float16* p = o3 + (size_t)tid * (32 * 16);
    float s = 0.0f;
    for (int i = 0; i < 512; ++i) s += (float)p[i];
    pooled[tid] = s * (1.0f / 512.0f);
}

// ---------------------------------------------------------------------------
// Kernel 4: key/query MLP towers + attention softmax. Single workgroup,
// everything LDS-resident (total work ~3 MFLOP).
// ---------------------------------------------------------------------------
__global__ void __launch_bounds__(256)
attention_kernel(const float* __restrict__ pooled,
                 const float* kf1w, const float* kf1b,
                 const float* kf2w, const float* kf2b,
                 const float* kf3w, const float* kf3b,
                 const float* qf1w, const float* qf1b,
                 const float* qf2w, const float* qf2b,
                 const float* qf3w, const float* qf3b,
                 const float* aw, const float* ab,
                 float* __restrict__ attn) {
    __shared__ float sp[NIMG * 64];
    __shared__ float h1[NIMG * 256];
    __shared__ float h2[NIMG * 128];
    __shared__ float sk[NIMG * 256];
    __shared__ float qq[B_ * 256];
    __shared__ float lg[NIMG];
    int t = threadIdx.x;

    for (int i = t; i < NIMG * 64; i += 256) sp[i] = pooled[i];
    __syncthreads();
    for (int i = t; i < NIMG * 256; i += 256) {       // keys fc1 (64->256) relu
        int r = i >> 8, j = i & 255;
        float s = kf1b[j];
        for (int k = 0; k < 64; ++k) s += sp[r * 64 + k] * kf1w[j * 64 + k];
        h1[i] = fmaxf(s, 0.0f);
    }
    __syncthreads();
    for (int i = t; i < NIMG * 128; i += 256) {       // keys fc2 (256->128) relu
        int r = i >> 7, j = i & 127;
        float s = kf2b[j];
        for (int k = 0; k < 256; ++k) s += h1[r * 256 + k] * kf2w[j * 256 + k];
        h2[i] = fmaxf(s, 0.0f);
    }
    __syncthreads();
    for (int i = t; i < NIMG * 256; i += 256) {       // keys fc3 (128->256)
        int r = i >> 8, j = i & 255;
        float s = kf3b[j];
        for (int k = 0; k < 128; ++k) s += h2[r * 128 + k] * kf3w[j * 128 + k];
        sk[i] = s;
    }
    __syncthreads();
    for (int i = t; i < B_ * 256; i += 256) {         // qry fc1 (rows n==0)
        int r = i >> 8, j = i & 255;
        float s = qf1b[j];
        for (int k = 0; k < 64; ++k) s += sp[(r * L_) * 64 + k] * qf1w[j * 64 + k];
        h1[i] = fmaxf(s, 0.0f);
    }
    __syncthreads();
    for (int i = t; i < B_ * 128; i += 256) {         // qry fc2
        int r = i >> 7, j = i & 127;
        float s = qf2b[j];
        for (int k = 0; k < 256; ++k) s += h1[r * 256 + k] * qf2w[j * 256 + k];
        h2[i] = fmaxf(s, 0.0f);
    }
    __syncthreads();
    for (int i = t; i < B_ * 32; i += 256) {          // qry fc3 (128->32)
        int r = i >> 5, j = i & 31;
        float s = qf3b[j];
        for (int k = 0; k < 128; ++k) s += h2[r * 128 + k] * qf3w[j * 128 + k];
        h1[i] = s;
    }
    __syncthreads();
    for (int i = t; i < B_ * 256; i += 256) {         // q = q3 @ aw.T + ab
        int r = i >> 8, j = i & 255;
        float s = ab[j];
        for (int k = 0; k < 32; ++k) s += h1[r * 32 + k] * aw[j * 32 + k];
        qq[i] = s;
    }
    __syncthreads();
    if (t < NIMG) {                                   // logits = keys . q
        int b = t / L_;
        float s = 0.0f;
        for (int k = 0; k < 256; ++k) s += sk[t * 256 + k] * qq[b * 256 + k];
        lg[t] = s;
    }
    __syncthreads();
    if (t < B_) {                                     // softmax over 5 agents
        float m = lg[t * L_];
        for (int n = 1; n < L_; ++n) m = fmaxf(m, lg[t * L_ + n]);
        float e[L_], den = 0.0f;
        for (int n = 0; n < L_; ++n) { e[n] = __expf(lg[t * L_ + n] - m); den += e[n]; }
        float inv = 1.0f / den;
        for (int n = 0; n < L_; ++n) attn[t * L_ + n] = e[n] * inv;
    }
}

// ---------------------------------------------------------------------------
// Kernel 5: fused output. Recompute bilinear sample from x in fp32 (full
// precision) and accumulate attention-weighted sum over the 5 agents.
// ---------------------------------------------------------------------------
__global__ void __launch_bounds__(256)
fusion_kernel(const float* __restrict__ x, const float* __restrict__ nam,
              const float* __restrict__ attn, float* __restrict__ out) {
    size_t tid = (size_t)blockIdx.x * 256 + threadIdx.x;  // B*64*H*W = 8388608
    int w = (int)(tid & (W_ - 1));
    size_t t = tid >> 7;
    int h = (int)(t & (H_ - 1));
    t >>= 8;
    int c = (int)(t & 63);
    int b = (int)(t >> 6);

    float gx = (w + 0.5f) * (2.0f / W_) - 1.0f;
    float gy = (h + 0.5f) * (2.0f / H_) - 1.0f;
    float acc = 0.0f;
    for (int n = 0; n < L_; ++n) {
        const float* th = nam + ((size_t)(b * L_ * L_) + n) * 6;
        float sx = th[0] * gx + th[1] * gy + th[2];
        float sy = th[3] * gx + th[4] * gy + th[5];
        float fx = ((sx + 1.0f) * W_ - 1.0f) * 0.5f;
        float fy = ((sy + 1.0f) * H_ - 1.0f) * 0.5f;
        float x0f = floorf(fx), y0f = floorf(fy);
        int x0 = (int)x0f, y0 = (int)y0f, x1 = x0 + 1, y1 = y0 + 1;
        float wx1 = fx - x0f, wx0 = 1.0f - wx1;
        float wy1 = fy - y0f, wy0 = 1.0f - wy1;
        float m00 = (x0 >= 0 && x0 < W_ && y0 >= 0 && y0 < H_) ? wx0 * wy0 : 0.0f;
        float m10 = (x1 >= 0 && x1 < W_ && y0 >= 0 && y0 < H_) ? wx1 * wy0 : 0.0f;
        float m01 = (x0 >= 0 && x0 < W_ && y1 >= 0 && y1 < H_) ? wx0 * wy1 : 0.0f;
        float m11 = (x1 >= 0 && x1 < W_ && y1 >= 0 && y1 < H_) ? wx1 * wy1 : 0.0f;
        int xc0 = x0 < 0 ? 0 : (x0 > W_ - 1 ? W_ - 1 : x0);
        int xc1 = x1 < 0 ? 0 : (x1 > W_ - 1 ? W_ - 1 : x1);
        int yc0 = y0 < 0 ? 0 : (y0 > H_ - 1 ? H_ - 1 : y0);
        int yc1 = y1 < 0 ? 0 : (y1 > H_ - 1 ? H_ - 1 : y1);
        const float* p = x + ((size_t)(b * L_ + n) * C_ + c) * H_ * W_;
        float v = m00 * p[yc0 * W_ + xc0] + m10 * p[yc0 * W_ + xc1] +
                  m01 * p[yc1 * W_ + xc0] + m11 * p[yc1 * W_ + xc1];
        acc += attn[b * L_ + n] * v;
    }
    out[tid] = acc;
}

// ---------------------------------------------------------------------------
// Launch: workspace layout (bytes):
//   neigh f16 : 0          .. 83,886,080
//   o1    f16 : 83,886,080 .. 94,371,840
//   o2    f16 : 94,371,840 .. 99,614,720
//   o3    f16 : 99,614,720 .. 100,925,440
//   pooled f32: 100,925,440 (+5120)
//   attn  f32 : 100,930,560 (+80)
// ---------------------------------------------------------------------------
extern "C" void kernel_launch(void* const* d_in, const int* in_sizes, int n_in,
                              void* d_out, int out_size, void* d_ws, size_t ws_size,
                              hipStream_t stream) {
    (void)in_sizes; (void)n_in; (void)out_size; (void)ws_size;
    const float* x    = (const float*)d_in[0];
    const float* nam  = (const float*)d_in[2];
    const float* w1   = (const float*)d_in[3];
    const float* b1   = (const float*)d_in[4];
    const float* w2   = (const float*)d_in[5];
    const float* b2   = (const float*)d_in[6];
    const float* w3   = (const float*)d_in[7];
    const float* b3   = (const float*)d_in[8];
    const float* kf1w = (const float*)d_in[9];
    const float* kf1b = (const float*)d_in[10];
    const float* kf2w = (const float*)d_in[11];
    const float* kf2b = (const float*)d_in[12];
    const float* kf3w = (const float*)d_in[13];
    const float* kf3b = (const float*)d_in[14];
    const float* qf1w = (const float*)d_in[15];
    const float* qf1b = (const float*)d_in[16];
    const float* qf2w = (const float*)d_in[17];
    const float* qf2b = (const float*)d_in[18];
    const float* qf3w = (const float*)d_in[19];
    const float* qf3b = (const float*)d_in[20];
    const float* aw   = (const float*)d_in[21];
    const float* ab   = (const float*)d_in[22];

    char* ws = (char*)d_ws;
    _Float16* neigh  = (_Float16*)(ws);
    _Float16* o1     = (_Float16*)(ws + 83886080ull);
    _Float16* o2     = (_Float16*)(ws + 94371840ull);
    _Float16* o3     = (_Float16*)(ws + 99614720ull);
    float*    pooled = (float*)   (ws + 100925440ull);
    float*    attn   = (float*)   (ws + 100930560ull);

    grid_sample_f16<<<2560, 256, 0, stream>>>(x, nam, neigh);

    // block tile = 32 cout x 64 spatial, 256 threads
    conv4x4s2_wmma<64, 32, 256, 128><<<dim3(128, 1, 20), 256, 0, stream>>>(neigh, w1, b1, o1);
    conv4x4s2_wmma<32, 64, 128, 64><<<dim3(32, 2, 20), 256, 0, stream>>>(o1, w2, b2, o2);
    conv4x4s2_wmma<64, 64, 64, 32><<<dim3(8, 2, 20), 256, 0, stream>>>(o2, w3, b3, o3);

    pool_mean<<<5, 256, 0, stream>>>(o3, pooled);

    attention_kernel<<<1, 256, 0, stream>>>(pooled, kf1w, kf1b, kf2w, kf2b,
                                            kf3w, kf3b, qf1w, qf1b, qf2w, qf2b,
                                            qf3w, qf3b, aw, ab, attn);

    fusion_kernel<<<32768, 256, 0, stream>>>(x, nam, attn, (float*)d_out);
}
